// BCAM_2370821947628
// MI455X (gfx1250) — matrix operand order repaired and checked by reading, not compiled, not run on hardware
//
#include <hip/hip_runtime.h>
#include <hip/hip_bf16.h>
#include <cmath>

// Problem constants
#define B_    32
#define HW_   900
#define HWP_  912            // HW padded to multiple of 16
#define HWPM_ 1024           // HW padded to multiple of 256 (M block tile) for batched GEMM
#define NL_   20
#define DIM_  512
#define VIN_  512
#define LIN_  768
#define MTOT_ (B_*HWP_)      // 29184 = 114 * 256
#define KREL_ 928            // rel-map K padded to multiple of 32
#define NSP_  960            // S gemm N padded to multiple of 64

typedef __attribute__((ext_vector_type(16))) __bf16 v16bf;
typedef __attribute__((ext_vector_type(8)))  __bf16 v8bf;
typedef __attribute__((ext_vector_type(8)))  float  v8f;

// ---------------------------------------------------------------------------
// WMMA fragment loaders (gfx1250 16-bit layouts, ISA 7.12.2) — unconditional;
// all operand buffers are padded so tile overreads stay inside allocations.
// A 16x32 (MxK): lane(h=lane>>4, m=lane&15): VGPR0-3 = K[k0+8h .. +8),
//                                            VGPR4-7 = K[k0+16+8h .. +8)
// B 32x16 (KxN): B[k][n] = W[n][k] (W row-major NxK): lane(h,n) holds
//                K[k0+8h .. +8) and K[k0+16+8h .. +8) from row n.
// ---------------------------------------------------------------------------
__device__ __forceinline__ v16bf load_a_frag(const __bf16* __restrict__ A, long lda,
                                             int mBase, int k0) {
  int lane = threadIdx.x & 31;
  const __bf16* p = A + (long)(mBase + (lane & 15)) * lda + k0 + 8 * (lane >> 4);
  v8bf lo = *(const v8bf*)(p);
  v8bf hi = *(const v8bf*)(p + 16);
  v16bf f;
#pragma unroll
  for (int i = 0; i < 8; ++i) { f[i] = lo[i]; f[i + 8] = hi[i]; }
  return f;
}

__device__ __forceinline__ v16bf load_b_frag(const __bf16* __restrict__ Bm, long ldb,
                                             int nBase, int k0) {
  int lane = threadIdx.x & 31;
  const __bf16* p = Bm + (long)(nBase + (lane & 15)) * ldb + k0 + 8 * (lane >> 4);
  v8bf lo = *(const v8bf*)(p);
  v8bf hi = *(const v8bf*)(p + 16);
  v16bf f;
#pragma unroll
  for (int i = 0; i < 8; ++i) { f[i] = lo[i]; f[i + 8] = hi[i]; }
  return f;
}

// ---------------------------------------------------------------------------
// Generic bf16 WMMA GEMM:  out = ACT( A1*B1^T [+ A2*B2^T] + bias1 [+ bias2] )
// 256 thr = 8 waves as 4(M)x2(N); wave tile 64x32 (4 A-frags x 2 B-frags =
// 8 WMMAs per 6 fragment loads); block tile 256x64.
// OUT: 0 f32 store; 1 bf16 store; 2 bf16 transposed-per-batch store (q3T);
//      3 f32 store into d_out skipping padded rows; 4 f32 accumulate into d_out.
// ACT: 0 none, 1 relu, 2 tanh.
// ---------------------------------------------------------------------------
template <int ACT, int OUT, bool DUAL>
__global__ __launch_bounds__(256) void gemm_kernel(
    const __bf16* __restrict__ A1, const __bf16* __restrict__ A2,
    const __bf16* __restrict__ B1, const __bf16* __restrict__ B2,
    const float* __restrict__ bias1, const float* __restrict__ bias2,
    float* __restrict__ outF, __bf16* __restrict__ outB,
    int N, int K, long lda, long ldb, long ldc,
    int Mlim, long bsA, long bsB, long bsC) {
  int z = blockIdx.z;
  A1 += (long)z * bsA;
  B1 += (long)z * bsB;
  if (outF) outF += (long)z * bsC;
  if (outB) outB += (long)z * bsC;

  int wave = threadIdx.x >> 5;
  int wm = wave >> 1, wn = wave & 1;
  int m0 = blockIdx.y * 256 + wm * 64;
  int n0 = blockIdx.x * 64 + wn * 32;

  v8f acc[4][2];
#pragma unroll
  for (int i = 0; i < 4; ++i)
#pragma unroll
    for (int j = 0; j < 2; ++j)
#pragma unroll
      for (int r = 0; r < 8; ++r) acc[i][j][r] = 0.f;

  for (int k0 = 0; k0 < K; k0 += 32) {
    // stream-ahead hint (speculative; dropped if past allocation)
    __builtin_prefetch((const void*)(A1 + (long)(m0 + (threadIdx.x & 31)) * lda + k0 + 32), 0, 1);
    v16bf a[4], b[2];
#pragma unroll
    for (int i = 0; i < 4; ++i) a[i] = load_a_frag(A1, lda, m0 + 16 * i, k0);
#pragma unroll
    for (int j = 0; j < 2; ++j) b[j] = load_b_frag(B1, ldb, n0 + 16 * j, k0);
#pragma unroll
    for (int i = 0; i < 4; ++i)
#pragma unroll
      for (int j = 0; j < 2; ++j)
        acc[i][j] = __builtin_amdgcn_wmma_f32_16x16x32_bf16(false, a[i], false, b[j],
                                                            (short)0, acc[i][j], false, false);
  }
  if (DUAL) {
    for (int k0 = 0; k0 < K; k0 += 32) {
      __builtin_prefetch((const void*)(A2 + (long)(m0 + (threadIdx.x & 31)) * lda + k0 + 32), 0, 1);
      v16bf a[4], b[2];
#pragma unroll
      for (int i = 0; i < 4; ++i) a[i] = load_a_frag(A2, lda, m0 + 16 * i, k0);
#pragma unroll
      for (int j = 0; j < 2; ++j) b[j] = load_b_frag(B2, ldb, n0 + 16 * j, k0);
#pragma unroll
      for (int i = 0; i < 4; ++i)
#pragma unroll
        for (int j = 0; j < 2; ++j)
          acc[i][j] = __builtin_amdgcn_wmma_f32_16x16x32_bf16(false, a[i], false, b[j],
                                                              (short)0, acc[i][j], false, false);
    }
  }

  // Epilogue. C/D layout: VGPR r -> M = r + 8h, N = lane&15.
  int lane = threadIdx.x & 31;
  int h = lane >> 4, nl = lane & 15;
#pragma unroll
  for (int i = 0; i < 4; ++i) {
#pragma unroll
    for (int j = 0; j < 2; ++j) {
      int n = n0 + j * 16 + nl;
      float bv = 0.f;
      if (bias1) bv += bias1[n];
      if (DUAL && bias2) bv += bias2[n];
#pragma unroll
      for (int r = 0; r < 8; ++r) {
        int m = m0 + i * 16 + r + 8 * h;
        if (m >= Mlim) continue;
        float v = acc[i][j][r] + bv;
        if (ACT == 1) v = fmaxf(v, 0.f);
        else if (ACT == 2) v = tanhf(v);
        if (OUT == 0) {
          outF[(long)m * ldc + n] = v;
        } else if (OUT == 1) {
          outB[(long)m * ldc + n] = (__bf16)v;
        } else if (OUT == 2) {  // transposed per batch: q3T[b][n][w], ldc = KREL_
          int b = m / HWP_, w = m % HWP_;
          outB[((long)b * DIM_ + n) * ldc + w] = (__bf16)v;
        } else if (OUT == 3) {  // store to d_out (B,900,512), drop padded rows
          int b = m / HWP_, mm = m % HWP_;
          if (mm < HW_) outF[((long)(b * HW_ + mm)) * ldc + n] = v;
        } else if (OUT == 4) {  // accumulate into d_out
          int b = m / HWP_, mm = m % HWP_;
          if (mm < HW_) outF[((long)(b * HW_ + mm)) * ldc + n] += v;
        }
      }
    }
  }
}

// ---------------------------------------------------------------------------
// Small helper kernels
// ---------------------------------------------------------------------------
__global__ void cvt_f32_bf16_kernel(const float* __restrict__ s, __bf16* __restrict__ d, long n) {
  long i = (long)blockIdx.x * blockDim.x + threadIdx.x;
  if (i < n) d[i] = (__bf16)s[i];
}

__global__ void zero_bf16_kernel(__bf16* __restrict__ d, long n) {
  long i = (long)blockIdx.x * blockDim.x + threadIdx.x;
  if (i < n) d[i] = (__bf16)0.f;
}

// zero rel pad rows [HWP_, HWPM_) per batch (read as A-operand padding)
__global__ void zero_rel_pad_kernel(__bf16* __restrict__ rel) {
  long i = (long)blockIdx.x * blockDim.x + threadIdx.x;
  const long perB = (long)(HWPM_ - HWP_) * KREL_;
  if (i >= (long)B_ * perB) return;
  long b = i / perB, rem = i % perB;
  long r = HWP_ + rem / KREL_, c = rem % KREL_;
  rel[((long)b * HWPM_ + r) * KREL_ + c] = (__bf16)0.f;
}

__global__ void cvt_pad_x_kernel(const float* __restrict__ x, __bf16* __restrict__ xb) {
  long i = (long)blockIdx.x * blockDim.x + threadIdx.x;
  long total = (long)B_ * HWP_ * VIN_;
  if (i >= total) return;
  long c = i % VIN_;
  long r = (i / VIN_) % HWP_;
  long b = i / ((long)VIN_ * HWP_);
  xb[i] = (r < HW_) ? (__bf16)x[((long)b * HW_ + r) * VIN_ + c] : (__bf16)0.f;
}

__global__ void pad_bias_a_kernel(const float* __restrict__ ba, float* __restrict__ bap) {
  int i = blockIdx.x * blockDim.x + threadIdx.x;
  if (i < NSP_) bap[i] = (i < HW_) ? ba[i] : 0.f;
}

// lt[b,n,d] = sum_c l[b,c,n] * W_lang[d,c] + b_lang[d]   (tiny: 0.5 GF)
__global__ void lang_reduce_kernel(const float* __restrict__ l, const float* __restrict__ W,
                                   const float* __restrict__ bias, float* __restrict__ lt) {
  int b = blockIdx.x / NL_, n = blockIdx.x % NL_;
  __shared__ float lcol[LIN_];
  for (int c = threadIdx.x; c < LIN_; c += blockDim.x)
    lcol[c] = l[(long)b * LIN_ * NL_ + (long)c * NL_ + n];
  __syncthreads();
  for (int d = threadIdx.x; d < DIM_; d += blockDim.x) {
    const float* w = W + (long)d * LIN_;
    float acc = bias[d];
    for (int c = 0; c < LIN_; ++c) acc = fmaf(lcol[c], w[c], acc);
    lt[((long)b * NL_ + n) * DIM_ + d] = acc;
  }
}

// Per-row NL=20 attention: sim -> masked softmax -> out = sim @ lt  (bf16 out)
__global__ __launch_bounds__(256) void attn_kernel(const __bf16* __restrict__ q1,
                                                   const float* __restrict__ lt,
                                                   const float* __restrict__ lmask,
                                                   __bf16* __restrict__ outb) {
  int row = blockIdx.x;              // 0 .. B_*HWP_-1
  int b = row / HWP_;
  __shared__ float sq[DIM_];
  __shared__ float part[NL_ * 8];
  __shared__ float prob[NL_];
  int t = threadIdx.x;
  sq[t]       = (float)q1[(long)row * DIM_ + t];
  sq[t + 256] = (float)q1[(long)row * DIM_ + t + 256];
  __syncthreads();
  if (t < NL_ * 8) {
    int n = t >> 3, ln = t & 7;
    const float* ltn = lt + ((long)b * NL_ + n) * DIM_;
    float s = 0.f;
    for (int d = ln; d < DIM_; d += 8) s = fmaf(sq[d], ltn[d], s);
    part[t] = s;
  }
  __syncthreads();
  if (t < NL_) {
    float s = 0.f;
#pragma unroll
    for (int i = 0; i < 8; ++i) s += part[t * 8 + i];
    prob[t] = s + 10000.f * lmask[b * NL_ + t] - 10000.f;
  }
  __syncthreads();
  if (t == 0) {
    float mx = prob[0];
    for (int n = 1; n < NL_; ++n) mx = fmaxf(mx, prob[n]);
    float sum = 0.f;
    for (int n = 0; n < NL_; ++n) { float e = __expf(prob[n] - mx); prob[n] = e; sum += e; }
    float inv = 1.f / sum;
    for (int n = 0; n < NL_; ++n) prob[n] *= inv;
  }
  __syncthreads();
  for (int d = t; d < DIM_; d += 256) {
    float o = 0.f;
#pragma unroll
    for (int n = 0; n < NL_; ++n) o = fmaf(prob[n], lt[((long)b * NL_ + n) * DIM_ + d], o);
    outb[(long)row * DIM_ + d] = (__bf16)o;
  }
}

// Row softmax over 900 logits (S ld = NSP_) -> rel bf16 (batch stride HWPM_ rows,
// ld KREL_, cols>=900 zero)
__global__ __launch_bounds__(256) void softmax900_kernel(const float* __restrict__ S,
                                                         __bf16* __restrict__ rel) {
  int row = blockIdx.x;              // 0 .. B_*HWP_-1
  int b = row / HWP_, rr = row % HWP_;
  const float* s = S + (long)row * NSP_;
  __bf16* r = rel + ((long)b * HWPM_ + rr) * KREL_;
  __shared__ float red[256];
  int t = threadIdx.x;
  float mx = -1e30f;
  for (int c = t; c < HW_; c += 256) mx = fmaxf(mx, s[c]);
  red[t] = mx; __syncthreads();
  for (int k = 128; k > 0; k >>= 1) { if (t < k) red[t] = fmaxf(red[t], red[t + k]); __syncthreads(); }
  mx = red[0]; __syncthreads();
  float sum = 0.f;
  for (int c = t; c < HW_; c += 256) sum += __expf(s[c] - mx);
  red[t] = sum; __syncthreads();
  for (int k = 128; k > 0; k >>= 1) { if (t < k) red[t] += red[t + k]; __syncthreads(); }
  float inv = 1.f / red[0];
  for (int c = t; c < KREL_; c += 256)
    r[c] = (c < HW_) ? (__bf16)(__expf(s[c] - mx) * inv) : (__bf16)0.f;
}

// ---------------------------------------------------------------------------
// Host driver
// ---------------------------------------------------------------------------
static inline long cdiv_l(long a, long b) { return (a + b - 1) / b; }

extern "C" void kernel_launch(void* const* d_in, const int* in_sizes, int n_in,
                              void* d_out, int out_size, void* d_ws, size_t ws_size,
                              hipStream_t stream) {
  const float* x      = (const float*)d_in[0];
  const float* l      = (const float*)d_in[1];
  const float* lmask  = (const float*)d_in[2];
  const float* W_lang = (const float*)d_in[3];
  const float* b_lang = (const float*)d_in[4];
  const float* W_v1   = (const float*)d_in[5];
  const float* b_v1   = (const float*)d_in[6];
  const float* W_v2   = (const float*)d_in[7];
  const float* b_v2   = (const float*)d_in[8];
  const float* W_v3   = (const float*)d_in[9];
  const float* b_v3   = (const float*)d_in[10];
  const float* W_v4   = (const float*)d_in[11];
  const float* b_v4   = (const float*)d_in[12];
  const float* W_out1 = (const float*)d_in[13];
  const float* b_out1 = (const float*)d_in[14];
  const float* W_v22  = (const float*)d_in[15];
  const float* b_v22  = (const float*)d_in[16];
  const float* W_a    = (const float*)d_in[17];
  const float* b_a    = (const float*)d_in[18];
  const float* W_o3   = (const float*)d_in[19];
  const float* b_o3   = (const float*)d_in[20];
  float* out = (float*)d_out;

  // Workspace carve-out (256B aligned)
  char* base = (char*)d_ws;
  size_t off = 0;
  auto alloc = [&](size_t bytes) -> char* {
    off = (off + 255) & ~(size_t)255;
    char* p = base + off;
    off += bytes;
    return p;
  };
  __bf16* xb     = (__bf16*)alloc((size_t)MTOT_ * VIN_ * 2);
  __bf16* wv1b   = (__bf16*)alloc((size_t)DIM_ * VIN_ * 2);
  __bf16* wv2b   = (__bf16*)alloc((size_t)DIM_ * VIN_ * 2);
  __bf16* wv3b   = (__bf16*)alloc((size_t)DIM_ * VIN_ * 2);
  __bf16* wv4b   = (__bf16*)alloc((size_t)DIM_ * VIN_ * 2);
  __bf16* wout1b = (__bf16*)alloc((size_t)DIM_ * DIM_ * 2);
  __bf16* wv22b  = (__bf16*)alloc((size_t)DIM_ * DIM_ * 2);
  __bf16* wab    = (__bf16*)alloc((size_t)NSP_ * DIM_ * 2);   // rows 900..959 zero
  __bf16* wo3b   = (__bf16*)alloc((size_t)DIM_ * 2 * DIM_ * 2);
  float*  bap    = (float*)alloc((size_t)NSP_ * 4);
  float*  lt     = (float*)alloc((size_t)B_ * NL_ * DIM_ * 4);
  __bf16* q1     = (__bf16*)alloc((size_t)MTOT_ * DIM_ * 2);
  __bf16* q2     = (__bf16*)alloc((size_t)MTOT_ * DIM_ * 2);
  __bf16* q3T    = (__bf16*)alloc((size_t)B_ * DIM_ * KREL_ * 2);
  __bf16* outb   = (__bf16*)alloc((size_t)MTOT_ * DIM_ * 2);
  __bf16* tanhA  = (__bf16*)alloc((size_t)MTOT_ * DIM_ * 2);
  float*  S      = (float*)alloc((size_t)MTOT_ * NSP_ * 4);
  __bf16* rel    = (__bf16*)alloc((size_t)B_ * HWPM_ * KREL_ * 2);  // rows 912..1023 zero
  __bf16* out2   = (__bf16*)alloc((size_t)MTOT_ * DIM_ * 2);
  (void)ws_size; (void)in_sizes; (void)n_in; (void)out_size;

  auto cvt = [&](const float* s, __bf16* d, long n) {
    cvt_f32_bf16_kernel<<<(unsigned)cdiv_l(n, 256), 256, 0, stream>>>(s, d, n);
  };

  // ---- conversions / padding ----
  {
    long n = (long)MTOT_ * VIN_;
    cvt_pad_x_kernel<<<(unsigned)cdiv_l(n, 256), 256, 0, stream>>>(x, xb);
  }
  cvt(W_v1, wv1b, (long)DIM_ * VIN_);
  cvt(W_v2, wv2b, (long)DIM_ * VIN_);
  cvt(W_v3, wv3b, (long)DIM_ * VIN_);
  cvt(W_v4, wv4b, (long)DIM_ * VIN_);
  cvt(W_out1, wout1b, (long)DIM_ * DIM_);
  cvt(W_v22, wv22b, (long)DIM_ * DIM_);
  cvt(W_a, wab, (long)HW_ * DIM_);
  zero_bf16_kernel<<<(unsigned)cdiv_l((long)(NSP_ - HW_) * DIM_, 256), 256, 0, stream>>>(
      wab + (long)HW_ * DIM_, (long)(NSP_ - HW_) * DIM_);
  cvt(W_o3, wo3b, (long)DIM_ * 2 * DIM_);
  pad_bias_a_kernel<<<(NSP_ + 255) / 256, 256, 0, stream>>>(b_a, bap);
  zero_bf16_kernel<<<(unsigned)cdiv_l((long)B_ * DIM_ * KREL_, 256), 256, 0, stream>>>(
      q3T, (long)B_ * DIM_ * KREL_);
  {
    long n = (long)B_ * (HWPM_ - HWP_) * KREL_;
    zero_rel_pad_kernel<<<(unsigned)cdiv_l(n, 256), 256, 0, stream>>>(rel);
  }

  // ---- lang reduce ----
  lang_reduce_kernel<<<B_ * NL_, 128, 0, stream>>>(l, W_lang, b_lang, lt);

  // ---- projections (WMMA) ----
  dim3 blk(256);
  dim3 gP(DIM_ / 64, MTOT_ / 256, 1);  // 8 x 114
  gemm_kernel<1, 1, false><<<gP, blk, 0, stream>>>(xb, nullptr, wv1b, nullptr, b_v1, nullptr,
      nullptr, q1, DIM_, VIN_, VIN_, VIN_, DIM_, MTOT_, 0, 0, 0);
  gemm_kernel<1, 1, false><<<gP, blk, 0, stream>>>(xb, nullptr, wv2b, nullptr, b_v2, nullptr,
      nullptr, q2, DIM_, VIN_, VIN_, VIN_, DIM_, MTOT_, 0, 0, 0);
  // q3 -> transposed store q3T[b][d][w], ldc = KREL_
  gemm_kernel<1, 2, false><<<gP, blk, 0, stream>>>(xb, nullptr, wv3b, nullptr, b_v3, nullptr,
      nullptr, q3T, DIM_, VIN_, VIN_, VIN_, KREL_, MTOT_, 0, 0, 0);
  // q4 -> stored directly into d_out (rows<900)
  gemm_kernel<1, 3, false><<<gP, blk, 0, stream>>>(xb, nullptr, wv4b, nullptr, b_v4, nullptr,
      out, nullptr, DIM_, VIN_, VIN_, VIN_, DIM_, MTOT_, 0, 0, 0);

  // ---- NL=20 attention -> out (bf16) ----
  attn_kernel<<<MTOT_, 256, 0, stream>>>(q1, lt, lmask, outb);

  // ---- tanh(out@W_out1^T + q2@W_v22^T + b) -> bf16 ----
  gemm_kernel<2, 1, true><<<gP, blk, 0, stream>>>(outb, q2, wout1b, wv22b, b_out1, b_v22,
      nullptr, tanhA, DIM_, DIM_, DIM_, DIM_, DIM_, MTOT_, 0, 0, 0);

  // ---- S = tanhA @ W_a^T + b_a -> f32 (N padded to 960, W_a rows padded/zeroed) ----
  dim3 gS(NSP_ / 64, MTOT_ / 256, 1);  // 15 x 114
  gemm_kernel<0, 0, false><<<gS, blk, 0, stream>>>(tanhA, nullptr, wab, nullptr, bap, nullptr,
      S, nullptr, NSP_, DIM_, DIM_, DIM_, NSP_, MTOT_, 0, 0, 0);

  // ---- rel = softmax900(S) -> bf16 ----
  softmax900_kernel<<<MTOT_, 256, 0, stream>>>(S, rel);

  // ---- out2 = rel @ q3 (batched): A=rel(b,1024,928), B=q3T(b,512,928) ----
  dim3 gO2(DIM_ / 64, HWPM_ / 256, B_);  // 8 x 4 x 32 (store-guarded at 912 rows)
  gemm_kernel<0, 1, false><<<gO2, blk, 0, stream>>>(rel, nullptr, q3T, nullptr, nullptr, nullptr,
      nullptr, out2, DIM_, KREL_, KREL_, KREL_, DIM_, HWP_,
      (long)HWPM_ * KREL_, (long)DIM_ * KREL_, (long)HWP_ * DIM_);

  // ---- d_out += relu([out2|out] @ W_o3^T + b_o3)  (d_out already holds q4) ----
  gemm_kernel<1, 4, true><<<gP, blk, 0, stream>>>(out2, outb, wo3b, wo3b + DIM_, b_o3, nullptr,
      out, nullptr, DIM_, DIM_, DIM_, 2 * DIM_, DIM_, MTOT_, 0, 0, 0);
}
